// MultiHeadAttentionWithSWA_4544075399592
// MI455X (gfx1250) — matrix operand order repaired and checked by reading, not compiled, not run on hardware
//
#include <hip/hip_runtime.h>
#include <stdint.h>
#include <stddef.h>

// ---------------------------------------------------------------------------
// Types / helpers
// ---------------------------------------------------------------------------
typedef __attribute__((ext_vector_type(16))) __bf16 bf16x16;
typedef __attribute__((ext_vector_type(8)))  float  f32x8;

union FragAB { bf16x16 v; uint32_t u[8]; };

static __device__ __forceinline__ uint16_t f2bf(float f) {
  // round-to-nearest-even f32 -> bf16 (inputs are finite/sane)
  uint32_t x = __float_as_uint(f);
  uint32_t r = (x + 0x7fffu + ((x >> 16) & 1u)) >> 16;
  return (uint16_t)r;
}

static __device__ __forceinline__ f32x8 wmma_bf16f32(const FragAB& a, const FragAB& b, f32x8 c) {
  return __builtin_amdgcn_wmma_f32_16x16x32_bf16(
      /*neg_a=*/false, a.v, /*neg_b=*/false, b.v,
      /*c_mod=*/(short)0, c, /*reuse_a=*/false, /*reuse_b=*/false);
}

// Problem constants
constexpr int S_LEN  = 4096;
constexpr int DMODEL = 1024;
constexpr int NHEAD  = 16;
constexpr int HDIM   = 64;
constexpr int WINSZ  = 512;

// GEMM tiling: 8 waves, wave grid 4x2, wave tile 32x64 -> block tile 128x128
constexpr int BM = 128;
constexpr int BN = 128;
constexpr int KT = 32;    // K step (one bf16 WMMA K)
constexpr int LDT = 34;   // padded LDS stride in bf16 (17 dwords, odd -> conflict-free)

// ---------------------------------------------------------------------------
// Generic bf16 WMMA GEMM: C[M,N] = A[M,K] * B[K,N] (+ bias)
// A is f32 or bf16 (converted to bf16 while staging to LDS); B is f32 weights.
// ---------------------------------------------------------------------------
template<bool A_IS_BF16, bool OUT_BF16, bool ADD_BIAS>
__global__ __launch_bounds__(256)
void gemm_bf16_wmma(const void* __restrict__ Ap, const float* __restrict__ Bw,
                    const float* __restrict__ bias, void* __restrict__ Cp,
                    int M, int N, int Kd) {
  __shared__ uint16_t As[BM * LDT];   // [row][k]
  __shared__ uint16_t Bs[BN * LDT];   // transposed: [n][k]

  const int tid   = threadIdx.x;
  const int lane  = tid & 31;
  const int wave  = tid >> 5;
  const int h16   = (lane >> 4) & 1;
  const int l15   = lane & 15;
  const int wrow  = wave >> 1;        // 0..3  -> rows wrow*32 .. +31
  const int wcol  = wave & 1;         // 0..1  -> cols wcol*64 .. +63
  const int m0    = blockIdx.y * BM;
  const int n0    = blockIdx.x * BN;

  f32x8 acc[2][4];
#pragma unroll
  for (int s = 0; s < 2; ++s)
#pragma unroll
    for (int t = 0; t < 4; ++t)
#pragma unroll
      for (int e = 0; e < 8; ++e) acc[s][t][e] = 0.0f;

  const uint32_t* As32 = (const uint32_t*)As;
  const uint32_t* Bs32 = (const uint32_t*)Bs;

  for (int k0 = 0; k0 < Kd; k0 += KT) {
    // ---- stage A tile (BM x KT) as bf16 ----
#pragma unroll
    for (int i = 0; i < (BM * KT) / 256; ++i) {
      int idx = i * 256 + tid;
      int row = idx >> 5;          // /32
      int col = idx & 31;
      if constexpr (A_IS_BF16) {
        const uint16_t* A16 = (const uint16_t*)Ap;
        As[row * LDT + col] = A16[(size_t)(m0 + row) * Kd + (k0 + col)];
      } else {
        const float* Af = (const float*)Ap;
        As[row * LDT + col] = f2bf(Af[(size_t)(m0 + row) * Kd + (k0 + col)]);
      }
    }
    // ---- stage B tile (KT x BN) transposed -> Bs[n][k] ----
#pragma unroll
    for (int i = 0; i < (BN * KT) / 256; ++i) {
      int idx = i * 256 + tid;
      int kr  = idx >> 7;          // /128 -> 0..31
      int nc  = idx & 127;
      Bs[nc * LDT + kr] = f2bf(Bw[(size_t)(k0 + kr) * N + (n0 + nc)]);
    }

    // ---- prefetch next K-step tiles into cache while this tile computes ----
    if (k0 + KT < Kd) {
      int kn = k0 + KT;
      {
        int row = tid >> 1;                       // 0..127
        int col = (tid & 1) * 16;
        if constexpr (A_IS_BF16) {
          const uint16_t* A16 = (const uint16_t*)Ap;
          __builtin_prefetch(&A16[(size_t)(m0 + row) * Kd + (kn + col)], 0, 0);
        } else {
          const float* Af = (const float*)Ap;
          __builtin_prefetch(&Af[(size_t)(m0 + row) * Kd + (kn + col)], 0, 0);
        }
      }
      {
        int kr = tid >> 3;                        // 0..31
        int nc = (tid & 7) * 16;
        __builtin_prefetch(&Bw[(size_t)(kn + kr) * N + (n0 + nc)], 0, 0);
      }
    }
    __syncthreads();

    // ---- A fragments (two 16x32 row sub-tiles) ----
    FragAB fa[2];
#pragma unroll
    for (int s = 0; s < 2; ++s) {
      int row = wrow * 32 + s * 16 + l15;
#pragma unroll
      for (int v = 0; v < 8; ++v) {
        int kp = v + (h16 ? 8 : 0);
        fa[s].u[v] = As32[row * 17 + kp];
      }
    }
    // ---- 4 B fragments (32x16), each reused for both row sub-tiles ----
#pragma unroll
    for (int t = 0; t < 4; ++t) {
      FragAB fb;
      int n = wcol * 64 + t * 16 + l15;
#pragma unroll
      for (int v = 0; v < 8; ++v) {
        int kp = v + (h16 ? 8 : 0);
        fb.u[v] = Bs32[n * 17 + kp];
      }
      acc[0][t] = wmma_bf16f32(fa[0], fb, acc[0][t]);
      acc[1][t] = wmma_bf16f32(fa[1], fb, acc[1][t]);
    }
    __syncthreads();
  }

  // ---- epilogue ----
#pragma unroll
  for (int s = 0; s < 2; ++s) {
#pragma unroll
    for (int t = 0; t < 4; ++t) {
#pragma unroll
      for (int r = 0; r < 8; ++r) {
        int row = m0 + wrow * 32 + s * 16 + r + (h16 ? 8 : 0);
        int col = n0 + wcol * 64 + t * 16 + l15;
        float vv = acc[s][t][r];
        if constexpr (ADD_BIAS) vv += bias[col];
        if constexpr (OUT_BF16)
          ((uint16_t*)Cp)[(size_t)row * N + col] = f2bf(vv);
        else
          ((float*)Cp)[(size_t)row * N + col] = vv;
      }
    }
  }
}

// ---------------------------------------------------------------------------
// Flash-style sliding-window attention. One wave handles one (head, 16-query)
// tile: S = Q K^T (masked, scaled), online softmax, ctx += P V.
// Q/K/V are bf16 [S, DMODEL] with head h occupying cols h*64 .. h*64+63.
// ---------------------------------------------------------------------------
__global__ __launch_bounds__(128)
void attn_swa_wmma(const uint16_t* __restrict__ Qg, const uint16_t* __restrict__ Kg,
                   const uint16_t* __restrict__ Vg, uint16_t* __restrict__ Ctx) {
  __shared__ uint16_t pst[4][16 * LDT];

  const int lane = threadIdx.x & 31;
  const int wv   = threadIdx.x >> 5;
  const int gw   = blockIdx.x * 4 + wv;
  const int head = gw >> 8;            // / 256 query-tiles
  const int qt   = gw & 255;
  const int q0   = qt * 16;
  const int hoff = head * HDIM;
  const int h16  = (lane >> 4) & 1;
  const int l15  = lane & 15;

  const uint32_t* Q32 = (const uint32_t*)Qg;
  const uint32_t* K32 = (const uint32_t*)Kg;
  constexpr int DW = DMODEL / 2;       // row stride in dwords

  // ---- Q fragments: two 16x32 slices covering d = 0..63 ----
  FragAB fq[2];
  {
    int row = q0 + l15;
#pragma unroll
    for (int s = 0; s < 2; ++s)
#pragma unroll
      for (int v = 0; v < 8; ++v) {
        int kp = v + (h16 ? 8 : 0);
        fq[s].u[v] = Q32[(size_t)row * DW + (hoff + s * 32) / 2 + kp];
      }
  }

  f32x8 acc[4];
  float mrow[8], lrow[8];
#pragma unroll
  for (int t = 0; t < 4; ++t)
#pragma unroll
    for (int e = 0; e < 8; ++e) acc[t][e] = 0.0f;
#pragma unroll
  for (int r = 0; r < 8; ++r) { mrow[r] = -__builtin_inff(); lrow[r] = 0.0f; }

  int kBeg = q0 - (WINSZ - 1); if (kBeg < 0) kBeg = 0;
  kBeg &= ~31;

  for (int kb = kBeg; kb <= q0 + 15; kb += 32) {
    // prefetch next key block's K and V rows
    {
      int pr = kb + 32 + lane;
      if (pr > S_LEN - 1) pr = S_LEN - 1;
      __builtin_prefetch(&Kg[(size_t)pr * DMODEL + hoff], 0, 0);
      __builtin_prefetch(&Vg[(size_t)pr * DMODEL + hoff], 0, 0);
    }

    // ---- scores for two 16-key groups ----
    f32x8 sc[2];
#pragma unroll
    for (int g = 0; g < 2; ++g) {
      int krow = kb + g * 16 + l15;
      if (krow > S_LEN - 1) krow = S_LEN - 1;
      FragAB fk0, fk1;
#pragma unroll
      for (int v = 0; v < 8; ++v) {
        int kp = v + (h16 ? 8 : 0);
        fk0.u[v] = K32[(size_t)krow * DW + (hoff +  0) / 2 + kp];
        fk1.u[v] = K32[(size_t)krow * DW + (hoff + 32) / 2 + kp];
      }
      f32x8 c;
#pragma unroll
      for (int e = 0; e < 8; ++e) c[e] = 0.0f;
      c = wmma_bf16f32(fq[0], fk0, c);
      c = wmma_bf16f32(fq[1], fk1, c);
      sc[g] = c;
    }

    // ---- mask, scale, online softmax ----
    float p0[8], p1[8], alpha[8];
#pragma unroll
    for (int r = 0; r < 8; ++r) {
      int q   = q0 + r + (h16 ? 8 : 0);
      int k0c = kb + l15;
      int k1c = kb + 16 + l15;
      bool v0 = (k0c <= q) && (q - k0c < WINSZ);
      bool v1 = (k1c <= q) && (q - k1c < WINSZ);
      float s0 = v0 ? sc[0][r] * 0.125f : -__builtin_inff();
      float s1 = v1 ? sc[1][r] * 0.125f : -__builtin_inff();
      float rm = fmaxf(s0, s1);
#pragma unroll
      for (int off = 1; off < 16; off <<= 1)
        rm = fmaxf(rm, __shfl_xor(rm, off, 32));
      float mold = mrow[r];
      float mnew = fmaxf(mold, rm);
      float a  = (mold == -__builtin_inff()) ? 0.0f : __expf(mold - mnew);
      float e0 = v0 ? __expf(s0 - mnew) : 0.0f;
      float e1 = v1 ? __expf(s1 - mnew) : 0.0f;
      float rs = e0 + e1;
#pragma unroll
      for (int off = 1; off < 16; off <<= 1)
        rs += __shfl_xor(rs, off, 32);
      mrow[r]  = mnew;
      lrow[r]  = lrow[r] * a + rs;
      alpha[r] = a;
      p0[r] = e0; p1[r] = e1;
    }
#pragma unroll
    for (int t = 0; t < 4; ++t)
#pragma unroll
      for (int r = 0; r < 8; ++r) acc[t][r] *= alpha[r];

    // ---- P (C layout) -> LDS -> A-fragment layout ----
    uint16_t* ps = pst[wv];
#pragma unroll
    for (int r = 0; r < 8; ++r) {
      int row = r + (h16 ? 8 : 0);
      ps[row * LDT + l15]      = f2bf(p0[r]);
      ps[row * LDT + 16 + l15] = f2bf(p1[r]);
    }
    asm volatile("s_wait_dscnt 0" ::: "memory");
    FragAB fp;
    {
      const uint32_t* ps32 = (const uint32_t*)ps;
      int row = l15;
#pragma unroll
      for (int v = 0; v < 8; ++v) {
        int kp = v + (h16 ? 8 : 0);
        fp.u[v] = ps32[row * 17 + kp];
      }
    }

    // ---- ctx += P * V  (4 d-tiles of 16) ----
#pragma unroll
    for (int t = 0; t < 4; ++t) {
      FragAB fv;
      int dcol = hoff + t * 16 + l15;
#pragma unroll
      for (int v = 0; v < 8; ++v) {
        int k0r = kb + 2 * v + (h16 ? 16 : 0);
        int k1r = k0r + 1;
        if (k0r > S_LEN - 1) k0r = S_LEN - 1;
        if (k1r > S_LEN - 1) k1r = S_LEN - 1;
        uint32_t lo = Vg[(size_t)k0r * DMODEL + dcol];
        uint32_t hi = Vg[(size_t)k1r * DMODEL + dcol];
        fv.u[v] = lo | (hi << 16);
      }
      acc[t] = wmma_bf16f32(fp, fv, acc[t]);
    }
  }

  // ---- normalize and store ctx (bf16) ----
#pragma unroll
  for (int t = 0; t < 4; ++t)
#pragma unroll
    for (int r = 0; r < 8; ++r) {
      int q    = q0 + r + (h16 ? 8 : 0);
      int dcol = hoff + t * 16 + l15;
      Ctx[(size_t)q * DMODEL + dcol] = f2bf(acc[t][r] / lrow[r]);
    }
}

// ---------------------------------------------------------------------------
// Launch
// ---------------------------------------------------------------------------
extern "C" void kernel_launch(void* const* d_in, const int* in_sizes, int n_in,
                              void* d_out, int out_size, void* d_ws, size_t ws_size,
                              hipStream_t stream) {
  const float* x  = (const float*)d_in[0];
  const float* Wq = (const float*)d_in[1];
  const float* Wk = (const float*)d_in[2];
  const float* Wv = (const float*)d_in[3];
  const float* Wo = (const float*)d_in[4];
  const float* bo = (const float*)d_in[5];

  const size_t elems = (size_t)S_LEN * DMODEL;
  uint16_t* Qb = (uint16_t*)d_ws;
  uint16_t* Kb = Qb + elems;
  uint16_t* Vb = Kb + elems;
  uint16_t* Cx = Vb + elems;

  dim3 gg(DMODEL / BN, S_LEN / BM);
  dim3 gb(256);

  // Q/K/V projections (f32 in, bf16 out)
  gemm_bf16_wmma<false, true, false><<<gg, gb, 0, stream>>>(x, Wq, nullptr, Qb, S_LEN, DMODEL, DMODEL);
  gemm_bf16_wmma<false, true, false><<<gg, gb, 0, stream>>>(x, Wk, nullptr, Kb, S_LEN, DMODEL, DMODEL);
  gemm_bf16_wmma<false, true, false><<<gg, gb, 0, stream>>>(x, Wv, nullptr, Vb, S_LEN, DMODEL, DMODEL);

  // sliding-window attention: 16 heads x 256 query tiles, 4 waves/block
  attn_swa_wmma<<<(NHEAD * 256) / 4, 128, 0, stream>>>(Qb, Kb, Vb, Cx);

  // output projection + bias (bf16 in, f32 out)
  gemm_bf16_wmma<true, false, true><<<gg, gb, 0, stream>>>(Cx, Wo, bo, d_out, S_LEN, DMODEL, DMODEL);
}
